// TemporalEdgeConv_7112465842373
// MI455X (gfx1250) — matrix operand order repaired and checked by reading, not compiled, not run on hardware
//
#include <hip/hip_runtime.h>

#define N_NODES 100000
#define N_EDGES 500000
#define IN_CH   128
#define EDGE_CH 64
#define NT32    (N_EDGES / 32)
#define SCR_LD  136   // padded f16 row stride (272B: 16B-aligned, bank-shifted)

typedef __attribute__((ext_vector_type(16))) _Float16 v16h;
typedef __attribute__((ext_vector_type(8)))  float    v8f;

union F16Frag { v16h v; uint4 q[2]; };

static __device__ __forceinline__ v8f wmma16(v16h a, v16h b, v8f c) {
  // (neg_a, A, neg_b, B, c_mod, C, reuse_a, reuse_b)
  return __builtin_amdgcn_wmma_f32_16x16x32_f16(false, a, false, b, (short)0, c, false, false);
}

static __device__ __forceinline__ v8f vzero() {
  v8f z;
#pragma unroll
  for (int i = 0; i < 8; ++i) z[i] = 0.0f;
  return z;
}

// A-fragment (16x32 f16) from a row-major fp32 row. Per ISA 7.12.2:
// lane holds row M=lane%16; VGPR0..3 = K(8*half + 0..7), VGPR4..7 = K(16 + 8*half + 0..7)
static __device__ __forceinline__ v16h a_frag_global(const float* __restrict__ row, int ks, int half) {
  const int c0 = ks * 32 + half * 8;
  float4 f0 = *(const float4*)(row + c0);
  float4 f1 = *(const float4*)(row + c0 + 4);
  float4 f2 = *(const float4*)(row + c0 + 16);
  float4 f3 = *(const float4*)(row + c0 + 20);
  F16Frag r;
  r.v[0]  = (_Float16)f0.x; r.v[1]  = (_Float16)f0.y; r.v[2]  = (_Float16)f0.z; r.v[3]  = (_Float16)f0.w;
  r.v[4]  = (_Float16)f1.x; r.v[5]  = (_Float16)f1.y; r.v[6]  = (_Float16)f1.z; r.v[7]  = (_Float16)f1.w;
  r.v[8]  = (_Float16)f2.x; r.v[9]  = (_Float16)f2.y; r.v[10] = (_Float16)f2.z; r.v[11] = (_Float16)f2.w;
  r.v[12] = (_Float16)f3.x; r.v[13] = (_Float16)f3.y; r.v[14] = (_Float16)f3.z; r.v[15] = (_Float16)f3.w;
  return r.v;
}

// A-fragment from a row-major f16 row (pre-converted x): two global b128 loads, no cvt.
static __device__ __forceinline__ v16h a_frag_f16row(const _Float16* __restrict__ row, int ks, int half) {
  const int c0 = ks * 32 + half * 8;
  F16Frag r;
  r.q[0] = *(const uint4*)(row + c0);
  r.q[1] = *(const uint4*)(row + c0 + 16);
  return r.v;
}

// A-fragment reload from the per-wave f16 transpose scratch: two ds_load_b128.
static __device__ __forceinline__ v16h a_frag_lds(const _Float16* scr, int m, int ks, int half) {
  const int c0 = ks * 32 + half * 8;
  F16Frag r;
  r.q[0] = *(const uint4*)(scr + m * SCR_LD + c0);
  r.q[1] = *(const uint4*)(scr + m * SCR_LD + c0 + 16);
  return r.v;
}

// B-fragment from pre-swizzled LDS: each lane reads 32 contiguous bytes.
static __device__ __forceinline__ v16h b_frag(const _Float16* w, int ks, int nt, int lane) {
  const _Float16* p = w + (((ks << 3) + nt) << 9) + (lane << 4);
  F16Frag r;
  r.q[0] = *(const uint4*)(p);
  r.q[1] = *(const uint4*)(p + 8);
  return r.v;
}

// Stage a row-major KxN(=128) fp32 weight into LDS as swizzled f16 B-fragments.
// B layout (ISA 7.12.2): lane holds col N=lane%16; elem j<8 = K(8*(lane/16) + j), elem j>=8 = K(16 + 8*(lane/16) + j-8).
static __device__ void stage_w(const float* __restrict__ W, int K, _Float16* dst, int tid, int nth) {
  const int total = (K / 32) * 8 * 32 * 16;
  for (int s = tid; s < total; s += nth) {
    int j  = s & 15;
    int l  = (s >> 4) & 31;
    int t  = s >> 9;
    int nt = t & 7;
    int ks = t >> 3;
    int kk = ks * 32 + (l >> 4) * 8 + (j & 7) + ((j >> 3) << 4);
    int nn = nt * 16 + (l & 15);
    dst[(t << 9) + (l << 4) + j] = (_Float16)W[kk * 128 + nn];
  }
}

__global__ void tec_init_out(const float4* __restrict__ x, float4* __restrict__ out, int n4) {
  int i = blockIdx.x * blockDim.x + threadIdx.x;
  if (i < n4) out[i] = x[i];
}

__global__ void tec_x_to_f16(const float* __restrict__ x, _Float16* __restrict__ xh, int n8) {
  int i = blockIdx.x * blockDim.x + threadIdx.x;
  if (i < n8) {
    float4 f0 = *(const float4*)(x + (size_t)i * 8);
    float4 f1 = *(const float4*)(x + (size_t)i * 8 + 4);
    union { uint4 q; _Float16 h[8]; } r;
    r.h[0] = (_Float16)f0.x; r.h[1] = (_Float16)f0.y; r.h[2] = (_Float16)f0.z; r.h[3] = (_Float16)f0.w;
    r.h[4] = (_Float16)f1.x; r.h[5] = (_Float16)f1.y; r.h[6] = (_Float16)f1.z; r.h[7] = (_Float16)f1.w;
    *(uint4*)(xh + (size_t)i * 8) = r.q;
  }
}

template <bool XF16>
__global__ __launch_bounds__(256, 1)
void tec_fused(const float* __restrict__ x, const _Float16* __restrict__ xh,
               const long long* __restrict__ ei,
               const float* __restrict__ ea, const float* __restrict__ ts,
               const float* __restrict__ We1, const float* __restrict__ be1,
               const float* __restrict__ We2, const float* __restrict__ be2,
               const float* __restrict__ Wt,  const float* __restrict__ bt,
               const float* __restrict__ Wn1, const float* __restrict__ bn1,
               const float* __restrict__ Wn2, const float* __restrict__ bn2,
               float* __restrict__ out) {
  extern __shared__ _Float16 smem[];
  _Float16* wWe1 = smem;                           // 8192 f16  (16 KB)
  _Float16* wWe2 = wWe1 + 8192;                    // 16384 f16 (32 KB)
  _Float16* wWn1 = wWe2 + 16384;                   // 32768 f16 (64 KB)
  _Float16* wWn2 = wWn1 + 32768;                   // 16384 f16 (32 KB)
  _Float16* scrb = wWn2 + 16384;                   // 8 waves * 32*SCR_LD f16

  const int tid = threadIdx.x;
  stage_w(We1, 64,  wWe1, tid, blockDim.x);
  stage_w(We2, 128, wWe2, tid, blockDim.x);
  stage_w(Wn1, 256, wWn1, tid, blockDim.x);
  stage_w(Wn2, 128, wWn2, tid, blockDim.x);
  __syncthreads();

  const int lane = tid & 31;
  const int half = (lane >> 4) & 1;
  const int ln   = lane & 15;
  _Float16* scr = scrb + (tid >> 5) * (32 * SCR_LD);

  // loop-invariant per-lane parameters (col n = 16*nt + ln in C layout)
  float be1r[8], be2r[8], bn1r[8], bn2r[8], wtr[8], btr[8];
#pragma unroll
  for (int nt = 0; nt < 8; ++nt) {
    int n = nt * 16 + ln;
    be1r[nt] = be1[n]; be2r[nt] = be2[n];
    bn1r[nt] = bn1[n]; bn2r[nt] = bn2[n];
    wtr[nt]  = Wt[n];  btr[nt]  = bt[n];
  }

  const long long* rowi = ei;
  const long long* coli = ei + N_EDGES;

  const int gw = (blockIdx.x * blockDim.x + tid) >> 5;
  const int nw = (gridDim.x * blockDim.x) >> 5;

  // each wave handles 32 edges (two 16-row M-tiles j=0,1) so every B-fragment
  // LDS load feeds two independent WMMAs
  for (int t = gw; t < NT32; t += nw) {
    const int e0 = t << 5;

    // prefetch next iteration's streaming operands (global_prefetch_b8)
    const int tn = t + nw;
    if (tn < NT32) {
      const int en = tn << 5;
      __builtin_prefetch(ea + (size_t)(en + ln) * EDGE_CH, 0, 0);
      __builtin_prefetch(ea + (size_t)(en + 16 + ln) * EDGE_CH, 0, 0);
      __builtin_prefetch(ts + en, 0, 0);
      __builtin_prefetch(rowi + en, 0, 0);
      __builtin_prefetch(coli + en, 0, 0);
    }

    v8f acc[2][8];

    // ---- phase 1: h = relu(edge_attr @ We1 + be1), K=64 ----
#pragma unroll
    for (int j = 0; j < 2; ++j)
#pragma unroll
      for (int nt = 0; nt < 8; ++nt) acc[j][nt] = vzero();
#pragma unroll
    for (int ks = 0; ks < 2; ++ks) {
      v16h aA = a_frag_global(ea + (size_t)(e0 + ln) * EDGE_CH, ks, half);
      v16h aB = a_frag_global(ea + (size_t)(e0 + 16 + ln) * EDGE_CH, ks, half);
#pragma unroll
      for (int nt = 0; nt < 8; ++nt) {
        v16h b = b_frag(wWe1, ks, nt, lane);
        acc[0][nt] = wmma16(aA, b, acc[0][nt]);
        acc[1][nt] = wmma16(aB, b, acc[1][nt]);
      }
    }
    // C layout: element v of acc[j][nt] is (row m = 16j + v + 8*half, col n = 16*nt + ln)
#pragma unroll
    for (int j = 0; j < 2; ++j)
#pragma unroll
      for (int nt = 0; nt < 8; ++nt)
#pragma unroll
        for (int v = 0; v < 8; ++v) {
          float h = acc[j][nt][v] + be1r[nt];
          scr[(j * 16 + v + 8 * half) * SCR_LD + nt * 16 + ln] = (_Float16)fmaxf(h, 0.0f);
        }

    // ---- phase 2: ef = (h @ We2 + be2) * sigmoid(ts*Wt + bt), K=128 ----
#pragma unroll
    for (int j = 0; j < 2; ++j)
#pragma unroll
      for (int nt = 0; nt < 8; ++nt) acc[j][nt] = vzero();
#pragma unroll
    for (int ks = 0; ks < 4; ++ks) {
      v16h aA = a_frag_lds(scr, ln, ks, half);
      v16h aB = a_frag_lds(scr + 16 * SCR_LD, ln, ks, half);
#pragma unroll
      for (int nt = 0; nt < 8; ++nt) {
        v16h b = b_frag(wWe2, ks, nt, lane);
        acc[0][nt] = wmma16(aA, b, acc[0][nt]);
        acc[1][nt] = wmma16(aB, b, acc[1][nt]);
      }
    }
    {
      float tsr[2][8];
#pragma unroll
      for (int j = 0; j < 2; ++j)
#pragma unroll
        for (int v = 0; v < 8; ++v) tsr[j][v] = ts[e0 + j * 16 + v + 8 * half];
#pragma unroll
      for (int j = 0; j < 2; ++j)
#pragma unroll
        for (int nt = 0; nt < 8; ++nt)
#pragma unroll
          for (int v = 0; v < 8; ++v) {
            float z  = tsr[j][v] * wtr[nt] + btr[nt];
            float tw = 1.0f / (1.0f + __expf(-z));
            float ef = (acc[j][nt][v] + be2r[nt]) * tw;
            scr[(j * 16 + v + 8 * half) * SCR_LD + nt * 16 + ln] = (_Float16)ef;
          }
    }

    // ---- phase 3: h2 = relu([x[row] | ef] @ Wn1 + bn1), K=256 ----
    const long long r0 = rowi[e0 + ln];
    const long long r1 = rowi[e0 + 16 + ln];
#pragma unroll
    for (int j = 0; j < 2; ++j)
#pragma unroll
      for (int nt = 0; nt < 8; ++nt) acc[j][nt] = vzero();
#pragma unroll
    for (int ks = 0; ks < 4; ++ks) {                 // top half: gathered x rows
      v16h aA, aB;
      if (XF16) {
        aA = a_frag_f16row(xh + (size_t)r0 * IN_CH, ks, half);
        aB = a_frag_f16row(xh + (size_t)r1 * IN_CH, ks, half);
      } else {
        aA = a_frag_global(x + (size_t)r0 * IN_CH, ks, half);
        aB = a_frag_global(x + (size_t)r1 * IN_CH, ks, half);
      }
#pragma unroll
      for (int nt = 0; nt < 8; ++nt) {
        v16h b = b_frag(wWn1, ks, nt, lane);
        acc[0][nt] = wmma16(aA, b, acc[0][nt]);
        acc[1][nt] = wmma16(aB, b, acc[1][nt]);
      }
    }
#pragma unroll
    for (int ks = 0; ks < 4; ++ks) {                 // bottom half: ef
      v16h aA = a_frag_lds(scr, ln, ks, half);
      v16h aB = a_frag_lds(scr + 16 * SCR_LD, ln, ks, half);
#pragma unroll
      for (int nt = 0; nt < 8; ++nt) {
        v16h b = b_frag(wWn1, ks + 4, nt, lane);
        acc[0][nt] = wmma16(aA, b, acc[0][nt]);
        acc[1][nt] = wmma16(aB, b, acc[1][nt]);
      }
    }
#pragma unroll
    for (int j = 0; j < 2; ++j)
#pragma unroll
      for (int nt = 0; nt < 8; ++nt)
#pragma unroll
        for (int v = 0; v < 8; ++v) {
          float h = acc[j][nt][v] + bn1r[nt];
          scr[(j * 16 + v + 8 * half) * SCR_LD + nt * 16 + ln] = (_Float16)fmaxf(h, 0.0f);
        }

    // ---- phase 4: msg = h2 @ Wn2 + bn2 ; scatter-add into out[col] ----
#pragma unroll
    for (int j = 0; j < 2; ++j)
#pragma unroll
      for (int nt = 0; nt < 8; ++nt) acc[j][nt] = vzero();
#pragma unroll
    for (int ks = 0; ks < 4; ++ks) {
      v16h aA = a_frag_lds(scr, ln, ks, half);
      v16h aB = a_frag_lds(scr + 16 * SCR_LD, ln, ks, half);
#pragma unroll
      for (int nt = 0; nt < 8; ++nt) {
        v16h b = b_frag(wWn2, ks, nt, lane);
        acc[0][nt] = wmma16(aA, b, acc[0][nt]);
        acc[1][nt] = wmma16(aB, b, acc[1][nt]);
      }
    }
#pragma unroll
    for (int j = 0; j < 2; ++j)
#pragma unroll
      for (int v = 0; v < 8; ++v) {
        const long long cdst = coli[e0 + j * 16 + v + 8 * half];
        float* op = out + (size_t)cdst * IN_CH + ln;
#pragma unroll
        for (int nt = 0; nt < 8; ++nt)
          atomicAdd(op + nt * 16, acc[j][nt][v] + bn2r[nt]);
      }
  }
}

extern "C" void kernel_launch(void* const* d_in, const int* in_sizes, int n_in,
                              void* d_out, int out_size, void* d_ws, size_t ws_size,
                              hipStream_t stream) {
  (void)in_sizes; (void)n_in; (void)out_size;
  const float*     x   = (const float*)d_in[0];
  const long long* ei  = (const long long*)d_in[1];   // int64 edge_index (2 x N_EDGES)
  const float*     ea  = (const float*)d_in[2];
  const float*     ts  = (const float*)d_in[3];
  const float*     We1 = (const float*)d_in[4];
  const float*     be1 = (const float*)d_in[5];
  const float*     We2 = (const float*)d_in[6];
  const float*     be2 = (const float*)d_in[7];
  const float*     Wt  = (const float*)d_in[8];
  const float*     bt  = (const float*)d_in[9];
  const float*     Wn1 = (const float*)d_in[10];
  const float*     bn1 = (const float*)d_in[11];
  const float*     Wn2 = (const float*)d_in[12];
  const float*     bn2 = (const float*)d_in[13];
  float* out = (float*)d_out;

  // pre-convert x to f16 in workspace (halves gather traffic; kills cvt in hot loop)
  const size_t xh_bytes = (size_t)N_NODES * IN_CH * sizeof(_Float16);
  const bool use_xf16 = ws_size >= xh_bytes;
  _Float16* xh = (_Float16*)d_ws;
  if (use_xf16) {
    const int n8 = (N_NODES * IN_CH) / 8;
    tec_x_to_f16<<<(n8 + 255) / 256, 256, 0, stream>>>(x, xh, n8);
  }

  // out = x (residual init), then edge kernel accumulates atomically on top.
  const int n4 = (N_NODES * IN_CH) / 4;
  tec_init_out<<<(n4 + 255) / 256, 256, 0, stream>>>((const float4*)x, (float4*)out, n4);

  // dynamic LDS: 144 KB swizzled f16 weights + 8 waves * 32*SCR_LD f16 scratch = 217,088 B
  const size_t lds_bytes = (size_t)(8192 + 16384 + 32768 + 16384 + 8 * 32 * SCR_LD) * sizeof(_Float16);
  if (use_xf16) {
    tec_fused<true><<<512, 256, lds_bytes, stream>>>(x, xh, ei, ea, ts, We1, be1, We2, be2,
                                                     Wt, bt, Wn1, bn1, Wn2, bn2, out);
  } else {
    tec_fused<false><<<512, 256, lds_bytes, stream>>>(x, nullptr, ei, ea, ts, We1, be1, We2, be2,
                                                      Wt, bt, Wn1, bn1, Wn2, bn2, out);
  }
}
